// GCN_24807731102257
// MI455X (gfx1250) — matrix operand order repaired and checked by reading, not compiled, not run on hardware
//
#include <hip/hip_runtime.h>
#include <hip/hip_bf16.h>
#include <stdint.h>

typedef __bf16 bf16_t;
typedef __attribute__((ext_vector_type(16))) __bf16 v16bf;
typedef __attribute__((ext_vector_type(8)))  float  v8f;
typedef __attribute__((ext_vector_type(4)))  uint32_t u32x4;
typedef __attribute__((ext_vector_type(8)))  uint32_t u32x8;

#define NND   10000
#define NE    320000
#define LM    512
#define G4    2048    // 4*LM
#define HID   256
#define OUTC  128
#define INPAD 32      // IN=26 padded to 32 for K%32==0

// ---------------------------------------------------------------------------
// elementwise helpers
// ---------------------------------------------------------------------------
__global__ void k_fill_f32(float* __restrict__ p, float v, long long n) {
  long long i = (long long)blockIdx.x * blockDim.x + threadIdx.x;
  if (i < n) p[i] = v;
}

__global__ void k_vec_add(const float* __restrict__ a, const float* __restrict__ b,
                          float* __restrict__ o, int n) {
  int i = blockIdx.x * blockDim.x + threadIdx.x;
  if (i < n) o[i] = a[i] + b[i];
}

// dst[r*stride + c] = (c<cols) ? bf16(src[r*cols+c]) : 0
__global__ void k_cast_pad_bf16(const float* __restrict__ src, bf16_t* __restrict__ dst,
                                int rows, int cols, int stride) {
  long long i = (long long)blockIdx.x * blockDim.x + threadIdx.x;
  long long total = (long long)rows * stride;
  if (i >= total) return;
  int r = (int)(i / stride), c = (int)(i % stride);
  dst[i] = (c < cols) ? (bf16_t)src[(long long)r * cols + c] : (bf16_t)0.0f;
}

// src is [K, Nn]; dst is [Nn, Kpad] with dst[c*Kpad + r] = src[r*Nn + c]
__global__ void k_cast_transpose_bf16(const float* __restrict__ src, bf16_t* __restrict__ dst,
                                      int K, int Nn, int Kpad) {
  long long i = (long long)blockIdx.x * blockDim.x + threadIdx.x;
  long long total = (long long)Nn * Kpad;
  if (i >= total) return;
  int c = (int)(i / Kpad), r = (int)(i % Kpad);
  dst[i] = (r < K) ? (bf16_t)src[(long long)r * Nn + c] : (bf16_t)0.0f;
}

// ---------------------------------------------------------------------------
// WMMA bf16 GEMM:  out[M,Nn] = act( A[M,K] @ Bt[Nn,K]^T + bias (+ addm) )
// A row-major bf16 (K%32==0), Bt row-major bf16 [Nn,K] (B transposed).
// One wave computes a 16x64 tile: one A fragment feeds 4 accumulators
// (4x v_wmma_f32_16x16x32_bf16 per K-step) for better arithmetic intensity.
// ---------------------------------------------------------------------------
__global__ void __launch_bounds__(256)
k_gemm_wmma_bf16(const bf16_t* __restrict__ A, const bf16_t* __restrict__ Bt,
                 const float* __restrict__ bias, const float* __restrict__ addm,
                 float* __restrict__ outf, bf16_t* __restrict__ outbf,
                 int M, int Nn, int K, int relu) {
  const int lane = threadIdx.x & 31;
  const int wave = threadIdx.x >> 5;
  const int ntN  = Nn >> 6;                // 64-wide column supertiles
  const int ntM  = M >> 4;
  const int tile = blockIdx.x * 8 + wave;
  if (tile >= ntM * ntN) return;           // wave-uniform exit
  const int tM = tile / ntN;
  const int tN = tile % ntN;
  const int hi = lane >> 4;                // 0 or 1 (lane half)
  const int lr = lane & 15;

  // fold the per-lane K sub-offsets into the base pointers
  const bf16_t* __restrict__ Ap = A  + (size_t)(tM * 16 + lr) * K + hi * 16;
  const bf16_t* __restrict__ Bp = Bt + (size_t)(tN * 64 + lr) * K + hi * 16;
  const size_t bstep = (size_t)16 * K;     // next 16-col B tile

  v8f acc0 = {0.f,0.f,0.f,0.f,0.f,0.f,0.f,0.f};
  v8f acc1 = acc0, acc2 = acc0, acc3 = acc0;

  union U { v16bf v; uint32_t u[8]; };

  for (int k0 = 0; k0 < K; k0 += 32) {
    U a, b0, b1, b2, b3;
#pragma unroll
    for (int j = 0; j < 8; ++j) {
      a.u[j]     = *(const uint32_t*)(Ap + k0 + 2 * j);
    }
#pragma unroll
    for (int j = 0; j < 8; ++j) b0.u[j] = *(const uint32_t*)(Bp + k0 + 2 * j);
#pragma unroll
    for (int j = 0; j < 8; ++j) b1.u[j] = *(const uint32_t*)(Bp + bstep + k0 + 2 * j);
#pragma unroll
    for (int j = 0; j < 8; ++j) b2.u[j] = *(const uint32_t*)(Bp + 2 * bstep + k0 + 2 * j);
#pragma unroll
    for (int j = 0; j < 8; ++j) b3.u[j] = *(const uint32_t*)(Bp + 3 * bstep + k0 + 2 * j);

    if (k0 + 128 < K) {                    // gfx1250 global_prefetch_b8
      __builtin_prefetch(Ap + k0 + 128, 0, 1);
      __builtin_prefetch(Bp + k0 + 128, 0, 1);
      __builtin_prefetch(Bp + bstep + k0 + 128, 0, 1);
      __builtin_prefetch(Bp + 2 * bstep + k0 + 128, 0, 1);
      __builtin_prefetch(Bp + 3 * bstep + k0 + 128, 0, 1);
    }

    acc0 = __builtin_amdgcn_wmma_f32_16x16x32_bf16(false, a.v, false, b0.v, (short)0, acc0, false, false);
    acc1 = __builtin_amdgcn_wmma_f32_16x16x32_bf16(false, a.v, false, b1.v, (short)0, acc1, false, false);
    acc2 = __builtin_amdgcn_wmma_f32_16x16x32_bf16(false, a.v, false, b2.v, (short)0, acc2, false, false);
    acc3 = __builtin_amdgcn_wmma_f32_16x16x32_bf16(false, a.v, false, b3.v, (short)0, acc3, false, false);
  }

  // C/D layout: lane l, vgpr v -> row = tM*16 + (l>=16?8:0)+v, col base + (l&15)
  const int r0 = tM * 16 + hi * 8;
  v8f accs[4] = {acc0, acc1, acc2, acc3};
#pragma unroll
  for (int n = 0; n < 4; ++n) {
    const int cc = tN * 64 + n * 16 + lr;
    const float bv = bias ? bias[cc] : 0.0f;
#pragma unroll
    for (int v = 0; v < 8; ++v) {
      size_t off = (size_t)(r0 + v) * Nn + cc;
      float val = accs[n][v] + bv;
      if (addm) val += addm[off];
      if (relu) val = fmaxf(val, 0.0f);
      if (outf)  outf[off]  = val;
      if (outbf) outbf[off] = (bf16_t)val;
    }
  }
}

// ---------------------------------------------------------------------------
// Tensor Data Mover: async DMA of one row of `nfloat` f32 from global to LDS.
// Builds the D# (group0 + group1, 2D tensor, groups 2/3 zero) in SGPRs and
// issues tensor_load_to_lds; completion tracked with TENSORcnt.
// ---------------------------------------------------------------------------
__device__ __forceinline__ uint32_t lds_offset_of(const void* p) {
  return (uint32_t)(uintptr_t)(__attribute__((address_space(3))) const void*)p;
}

__device__ __forceinline__ void tdm_load_row_f32(const float* gptr, uint32_t lds_off,
                                                 uint32_t nfloat) {
  uint64_t ga = (uint64_t)(uintptr_t)gptr;
  u32x4 g0;
  g0[0] = 1u;                                              // count=1 (valid user D#)
  g0[1] = lds_off;                                         // lds_addr (bytes)
  g0[2] = (uint32_t)ga;                                    // global_addr[31:0]
  g0[3] = (uint32_t)((ga >> 32) & 0x1FFFFFFu) | (2u << 30); // addr[56:32] | type=2
  u32x8 g1;
  g1[0] = 2u << 16;                                        // data_size=4B, no multicast
  g1[1] = (nfloat & 0xFFFFu) << 16;                        // tensor_dim0[15:0]
  g1[2] = ((nfloat >> 16) & 0xFFFFu) | (1u << 16);         // tensor_dim0[31:16], tensor_dim1=1
  g1[3] = (nfloat & 0xFFFFu) << 16;                        // tensor_dim1 hi=0, tile_dim0=nfloat
  g1[4] = 1u;                                              // tile_dim1=1, tile_dim2=0
  g1[5] = nfloat;                                          // tensor_dim0_stride[31:0]
  g1[6] = 0u;                                              // dim0_stride hi, dim1_stride lo
  g1[7] = 0u;
  u32x4 gz = {0u, 0u, 0u, 0u};
  asm volatile("tensor_load_to_lds %0, %1, %2, %3"
               :: "s"(g0), "s"(g1), "s"(gz), "s"(gz)
               : "memory");
}

// ---------------------------------------------------------------------------
// Persistent single-workgroup LSTM recurrence (torch gate order i,f,g,o).
// xg[T, 2048] precomputed; Whh bf16 [2048, 512] (L2-resident).
// xg rows are double-buffered in LDS via the Tensor Data Mover so the fetch
// of step t+1 hides behind the matvec of step t.
// ---------------------------------------------------------------------------
__global__ void __launch_bounds__(1024)
k_lstm_recurrence(const float* __restrict__ xg, const bf16_t* __restrict__ Whh,
                  bf16_t* __restrict__ hout, int T) {
  __shared__ float h[LM];
  __shared__ float c[LM];
  __shared__ float gates[G4];
  __shared__ float xbuf[2][G4];
  const int tid  = threadIdx.x;
  const int wv   = tid >> 5;
  if (tid < LM) { h[tid] = 0.0f; c[tid] = 0.0f; }
  if (wv == 0) {                                  // wave 0 drives the TDM
    tdm_load_row_f32(xg, lds_offset_of(&xbuf[0][0]), G4);
    __builtin_amdgcn_s_wait_tensorcnt(0);
  }
  __syncthreads();

  for (int t = 0; t < T; ++t) {
    const float* __restrict__ cur = xbuf[t & 1];
    if (wv == 0 && (t + 1) < T)
      tdm_load_row_f32(xg + (size_t)(t + 1) * G4,
                       lds_offset_of(&xbuf[(t + 1) & 1][0]), G4);

    const int j0 = tid * 2;
    const bf16_t* __restrict__ w0 = Whh + (size_t)j0 * LM;
    const bf16_t* __restrict__ w1 = w0 + LM;
    float a0 = cur[j0], a1 = cur[j0 + 1];
#pragma unroll 8
    for (int k = 0; k < LM; ++k) {
      float hv = h[k];
      a0 = fmaf((float)w0[k], hv, a0);
      a1 = fmaf((float)w1[k], hv, a1);
    }
    gates[j0] = a0;
    gates[j0 + 1] = a1;
    __syncthreads();
    if (tid < LM) {
      float ig = 1.0f / (1.0f + __expf(-gates[tid]));
      float fg = 1.0f / (1.0f + __expf(-gates[LM + tid]));
      float g  = tanhf(gates[2 * LM + tid]);
      float og = 1.0f / (1.0f + __expf(-gates[3 * LM + tid]));
      float cn = fg * c[tid] + ig * g;
      c[tid] = cn;
      float hn = og * tanhf(cn);
      h[tid] = hn;
      hout[(size_t)t * LM + tid] = (bf16_t)hn;
    }
    if (wv == 0 && (t + 1) < T)
      __builtin_amdgcn_s_wait_tensorcnt(0);       // next xg row landed in LDS
    __syncthreads();
  }
}

// ---------------------------------------------------------------------------
// GCN normalization + aggregation
// ---------------------------------------------------------------------------
__global__ void k_deg(const long long* __restrict__ ei, const float* __restrict__ ew,
                      float* __restrict__ deg, int E) {
  int e = blockIdx.x * blockDim.x + threadIdx.x;
  if (e >= E) return;
  atomicAdd(&deg[ei[E + e]], ew[e]);      // dst-side degree
}

__global__ void k_dinv(const float* __restrict__ deg, float* __restrict__ dinv,
                       float* __restrict__ selfnorm, int n) {
  int i = blockIdx.x * blockDim.x + threadIdx.x;
  if (i >= n) return;
  float d = deg[i];
  float r = (d > 0.0f) ? rsqrtf(d) : 0.0f;
  dinv[i] = r;
  selfnorm[i] = r * r;
}

__global__ void k_norm(const long long* __restrict__ ei, const float* __restrict__ ew,
                       const float* __restrict__ dinv, float* __restrict__ norm, int E) {
  int e = blockIdx.x * blockDim.x + threadIdx.x;
  if (e >= E) return;
  norm[e] = dinv[ei[e]] * ew[e] * dinv[ei[E + e]];
}

// one block per edge, one lane per column
__global__ void k_agg(const float* __restrict__ h, const float* __restrict__ norm,
                      const long long* __restrict__ ei, float* __restrict__ agg,
                      int E, int C) {
  int e = blockIdx.x;
  int c = threadIdx.x;
  long long s = ei[e], d = ei[E + e];
  float nv = norm[e];
  atomicAdd(&agg[(size_t)d * C + c], h[(size_t)s * C + c] * nv);
}

__global__ void k_finish(const float* __restrict__ agg, const float* __restrict__ h,
                         const float* __restrict__ selfnorm, const float* __restrict__ bias,
                         float* __restrict__ outf, bf16_t* __restrict__ outbf,
                         int n, int C, int relu) {
  long long i = (long long)blockIdx.x * blockDim.x + threadIdx.x;
  long long total = (long long)n * C;
  if (i >= total) return;
  int node = (int)(i / C), c = (int)(i % C);
  float val = agg[i] + h[i] * selfnorm[node] + bias[c];
  if (relu) val = fmaxf(val, 0.0f);
  if (outf)  outf[i]  = val;
  if (outbf) outbf[i] = (bf16_t)val;
}

// ---------------------------------------------------------------------------
// host orchestration
// ---------------------------------------------------------------------------
static inline void launch_gemm(const bf16_t* A, const bf16_t* Bt, const float* bias,
                               const float* addm, float* outf, bf16_t* outbf,
                               int M, int Nn, int K, int relu, hipStream_t s) {
  int tiles = (M / 16) * (Nn / 64);
  int blocks = (tiles + 7) / 8;
  k_gemm_wmma_bf16<<<blocks, 256, 0, s>>>(A, Bt, bias, addm, outf, outbf, M, Nn, K, relu);
}

static inline long long cdiv(long long a, long long b) { return (a + b - 1) / b; }

extern "C" void kernel_launch(void* const* d_in, const int* in_sizes, int n_in,
                              void* d_out, int out_size, void* d_ws, size_t ws_size,
                              hipStream_t stream) {
  (void)in_sizes; (void)n_in; (void)out_size; (void)ws_size;
  const float*     x    = (const float*)d_in[0];
  const long long* ei   = (const long long*)d_in[1];
  const float*     ew   = (const float*)d_in[2];
  const float*     aaW  = (const float*)d_in[3];
  const float*     lmW  = (const float*)d_in[4];
  const float*     lmb  = (const float*)d_in[5];
  const float*     Wih0 = (const float*)d_in[6];
  const float*     Whh0 = (const float*)d_in[7];
  const float*     bih0 = (const float*)d_in[8];
  const float*     bhh0 = (const float*)d_in[9];
  const float*     Wih1 = (const float*)d_in[10];
  const float*     Whh1 = (const float*)d_in[11];
  const float*     bih1 = (const float*)d_in[12];
  const float*     bhh1 = (const float*)d_in[13];
  const float*     W1   = (const float*)d_in[14];
  const float*     b1   = (const float*)d_in[15];
  const float*     W2   = (const float*)d_in[16];
  const float*     b2   = (const float*)d_in[17];
  const float*     W3   = (const float*)d_in[18];
  const float*     b3   = (const float*)d_in[19];
  float* out = (float*)d_out;

  // bump allocator in d_ws
  char* wsbase = (char*)d_ws;
  size_t off = 0;
  auto alloc = [&](size_t bytes) -> char* {
    char* p = wsbase + off;
    off += (bytes + 255) & ~(size_t)255;
    return p;
  };

  bf16_t* xbf    = (bf16_t*)alloc((size_t)NND * INPAD * 2);
  bf16_t* Wih0bf = (bf16_t*)alloc((size_t)G4 * INPAD * 2);
  bf16_t* Whh0bf = (bf16_t*)alloc((size_t)G4 * LM * 2);
  bf16_t* Wih1bf = (bf16_t*)alloc((size_t)G4 * LM * 2);
  bf16_t* Whh1bf = (bf16_t*)alloc((size_t)G4 * LM * 2);
  bf16_t* aaWt   = (bf16_t*)alloc((size_t)LM * INPAD * 2);
  bf16_t* lmWt   = (bf16_t*)alloc((size_t)LM * LM * 2);
  bf16_t* W1t    = (bf16_t*)alloc((size_t)HID * LM * 2);
  bf16_t* W2t    = (bf16_t*)alloc((size_t)HID * HID * 2);
  bf16_t* W3t    = (bf16_t*)alloc((size_t)OUTC * HID * 2);
  float*  b01    = (float*)alloc((size_t)G4 * 4);
  float*  b11    = (float*)alloc((size_t)G4 * 4);
  float*  xg     = (float*)alloc((size_t)NND * G4 * 4);
  bf16_t* h0bf   = (bf16_t*)alloc((size_t)NND * LM * 2);
  bf16_t* h1bf   = (bf16_t*)alloc((size_t)NND * LM * 2);
  float*  tmpf   = (float*)alloc((size_t)NND * LM * 4);
  bf16_t* zbf    = (bf16_t*)alloc((size_t)NND * LM * 2);
  float*  deg    = (float*)alloc((size_t)NND * 4);
  float*  dinv   = (float*)alloc((size_t)NND * 4);
  float*  selfn  = (float*)alloc((size_t)NND * 4);
  float*  normv  = (float*)alloc((size_t)NE * 4);
  float*  hbuf   = (float*)alloc((size_t)NND * HID * 4);
  float*  agg    = (float*)alloc((size_t)NND * HID * 4);
  bf16_t* abf    = (bf16_t*)alloc((size_t)NND * HID * 2);

  const int TPB = 256;
  // ---- weight / input casts (bf16, pre-transposed to [N,K], K padded) ----
  k_cast_pad_bf16<<<cdiv((long long)NND * INPAD, TPB), TPB, 0, stream>>>(x, xbf, NND, 26, INPAD);
  k_cast_pad_bf16<<<cdiv((long long)G4 * INPAD, TPB), TPB, 0, stream>>>(Wih0, Wih0bf, G4, 26, INPAD);
  k_cast_pad_bf16<<<cdiv((long long)G4 * LM, TPB), TPB, 0, stream>>>(Whh0, Whh0bf, G4, LM, LM);
  k_cast_pad_bf16<<<cdiv((long long)G4 * LM, TPB), TPB, 0, stream>>>(Wih1, Wih1bf, G4, LM, LM);
  k_cast_pad_bf16<<<cdiv((long long)G4 * LM, TPB), TPB, 0, stream>>>(Whh1, Whh1bf, G4, LM, LM);
  k_cast_transpose_bf16<<<cdiv((long long)LM * INPAD, TPB), TPB, 0, stream>>>(aaW, aaWt, 26, LM, INPAD);
  k_cast_transpose_bf16<<<cdiv((long long)LM * LM, TPB), TPB, 0, stream>>>(lmW, lmWt, LM, LM, LM);
  k_cast_transpose_bf16<<<cdiv((long long)HID * LM, TPB), TPB, 0, stream>>>(W1, W1t, LM, HID, LM);
  k_cast_transpose_bf16<<<cdiv((long long)HID * HID, TPB), TPB, 0, stream>>>(W2, W2t, HID, HID, HID);
  k_cast_transpose_bf16<<<cdiv((long long)OUTC * HID, TPB), TPB, 0, stream>>>(W3, W3t, HID, OUTC, HID);
  k_vec_add<<<cdiv(G4, TPB), TPB, 0, stream>>>(bih0, bhh0, b01, G4);
  k_vec_add<<<cdiv(G4, TPB), TPB, 0, stream>>>(bih1, bhh1, b11, G4);

  // ---- LSTM layer 0 ----
  launch_gemm(xbf, Wih0bf, b01, nullptr, xg, nullptr, NND, G4, INPAD, 0, stream);
  k_lstm_recurrence<<<1, 1024, 0, stream>>>(xg, Whh0bf, h0bf, NND);
  // ---- LSTM layer 1 ----
  launch_gemm(h0bf, Wih1bf, b11, nullptr, xg, nullptr, NND, G4, LM, 0, stream);
  k_lstm_recurrence<<<1, 1024, 0, stream>>>(xg, Whh1bf, h1bf, NND);

  // ---- z = relu(x@aa_W + h1@lm_W + lm_b) ----
  launch_gemm(h1bf, lmWt, lmb, nullptr, tmpf, nullptr, NND, LM, LM, 0, stream);
  launch_gemm(xbf, aaWt, nullptr, tmpf, tmpf, zbf, NND, LM, INPAD, 1, stream);

  // ---- GCN normalization (shared across all three convs) ----
  k_fill_f32<<<cdiv(NND, TPB), TPB, 0, stream>>>(deg, 1.0f, NND);   // self-loop weight 1
  k_deg<<<cdiv(NE, TPB), TPB, 0, stream>>>(ei, ew, deg, NE);
  k_dinv<<<cdiv(NND, TPB), TPB, 0, stream>>>(deg, dinv, selfn, NND);
  k_norm<<<cdiv(NE, TPB), TPB, 0, stream>>>(ei, ew, dinv, normv, NE);

  // ---- conv1: 512 -> 256, relu ----
  launch_gemm(zbf, W1t, nullptr, nullptr, hbuf, nullptr, NND, HID, LM, 0, stream);
  k_fill_f32<<<cdiv((long long)NND * HID, TPB), TPB, 0, stream>>>(agg, 0.0f, (long long)NND * HID);
  k_agg<<<NE, HID, 0, stream>>>(hbuf, normv, ei, agg, NE, HID);
  k_finish<<<cdiv((long long)NND * HID, TPB), TPB, 0, stream>>>(agg, hbuf, selfn, b1,
                                                               nullptr, abf, NND, HID, 1);
  // ---- conv2: 256 -> 256, relu ----
  launch_gemm(abf, W2t, nullptr, nullptr, hbuf, nullptr, NND, HID, HID, 0, stream);
  k_fill_f32<<<cdiv((long long)NND * HID, TPB), TPB, 0, stream>>>(agg, 0.0f, (long long)NND * HID);
  k_agg<<<NE, HID, 0, stream>>>(hbuf, normv, ei, agg, NE, HID);
  k_finish<<<cdiv((long long)NND * HID, TPB), TPB, 0, stream>>>(agg, hbuf, selfn, b2,
                                                               nullptr, abf, NND, HID, 1);
  // ---- conv3: 256 -> 128, no relu, into d_out ----
  launch_gemm(abf, W3t, nullptr, nullptr, hbuf, nullptr, NND, OUTC, HID, 0, stream);
  k_fill_f32<<<cdiv((long long)NND * OUTC, TPB), TPB, 0, stream>>>(agg, 0.0f, (long long)NND * OUTC);
  k_agg<<<NE, OUTC, 0, stream>>>(hbuf, normv, ei, agg, NE, OUTC);
  k_finish<<<cdiv((long long)NND * OUTC, TPB), TPB, 0, stream>>>(agg, hbuf, selfn, b3,
                                                                out, nullptr, NND, OUTC, 0);
}
